// MultiHeadAttention_23811298689173
// MI455X (gfx1250) — matrix operand order repaired and checked by reading, not compiled, not run on hardware
//
#include <hip/hip_runtime.h>
#include <hip/hip_bf16.h>

// ---------- types ----------
typedef __bf16 bf16;
typedef __attribute__((ext_vector_type(16))) __bf16 v16bf;
typedef __attribute__((ext_vector_type(8)))  __bf16 v8bf;
typedef __attribute__((ext_vector_type(8)))  float  v8f;
typedef __attribute__((ext_vector_type(4)))  unsigned int u32x4;
typedef __attribute__((ext_vector_type(8)))  int i32x8;
typedef __attribute__((ext_vector_type(4)))  int i32x4;

union V16 { v16bf v; v8bf h[2]; };

__device__ __forceinline__ v8f vzero8() {
    v8f z = {0.f,0.f,0.f,0.f,0.f,0.f,0.f,0.f};
    return z;
}

__device__ __forceinline__ bf16 f2bf(float f) {
    unsigned u = __float_as_uint(f);
    unsigned r = (u + 0x7FFFu + ((u >> 16) & 1u)) >> 16;   // RNE
    unsigned short s = (unsigned short)r;
    return __builtin_bit_cast(bf16, s);
}

__device__ __forceinline__ v8f wmma_bf16(v16bf a, v16bf b, v8f c) {
    return __builtin_amdgcn_wmma_f32_16x16x32_bf16(
        false, a, false, b, (short)0, c, false, false);
}

// ---------- DPP 16-lane xor-reductions (stay within each 16-lane row) ----------
template <int CTRL>
__device__ __forceinline__ float dpp_mov(float x) {
    return __builtin_bit_cast(float,
        __builtin_amdgcn_update_dpp(0, __builtin_bit_cast(int, x),
                                    CTRL, 0xF, 0xF, true));
}
__device__ __forceinline__ float red_max16(float x) {
    x = fmaxf(x, dpp_mov<0xB1>(x));    // quad_perm(1,0,3,2)  : xor 1
    x = fmaxf(x, dpp_mov<0x4E>(x));    // quad_perm(2,3,0,1)  : xor 2
    x = fmaxf(x, dpp_mov<0x141>(x));   // row_half_mirror     : quad ^ 1
    x = fmaxf(x, dpp_mov<0x140>(x));   // row_mirror          : quad ^ 3
    return x;
}
__device__ __forceinline__ float red_sum16(float x) {
    x += dpp_mov<0xB1>(x);
    x += dpp_mov<0x4E>(x);
    x += dpp_mov<0x141>(x);
    x += dpp_mov<0x140>(x);
    return x;
}

// ---------- Tensor Data Mover: 2D tile global -> LDS (D# per ISA 08 §8.3/8.4) ----------
__device__ __forceinline__ void tdm_load_2d(unsigned lds_off, const void* gaddr,
                                            unsigned tensor_d0, unsigned tensor_d1,
                                            unsigned tile_d0, unsigned tile_d1,
                                            unsigned stride_d0) {
    unsigned long long ga = (unsigned long long)(uintptr_t)gaddr;
    u32x4 g0;
    g0[0] = 1u;                                                  // count=1, user mode
    g0[1] = lds_off;                                             // lds_addr
    g0[2] = (unsigned)ga;                                        // global_addr[31:0]
    g0[3] = ((unsigned)(ga >> 32) & 0x01FFFFFFu) | 0x80000000u;  // addr[56:32] | type=2
    i32x8 g1;
    g1[0] = 0x00010000;                                          // data_size = 2 bytes
    g1[1] = (int)((tensor_d0 & 0xFFFFu) << 16);                  // tensor_dim0 lo16
    g1[2] = (int)(((tensor_d0 >> 16) & 0xFFFFu) |
                  ((tensor_d1 & 0xFFFFu) << 16));                // dim0 hi | dim1 lo
    g1[3] = (int)(((tensor_d1 >> 16) & 0xFFFFu) |
                  (tile_d0 << 16));                              // dim1 hi | tile_dim0
    g1[4] = (int)tile_d1;                                        // tile_dim1 (tile_dim2=0)
    g1[5] = (int)stride_d0;                                      // tensor_dim0_stride lo32
    g1[6] = 0;
    g1[7] = 0;
    i32x4 z4 = {0, 0, 0, 0};
    i32x8 z8 = {0, 0, 0, 0, 0, 0, 0, 0};
    __builtin_amdgcn_tensor_load_to_lds(g0, g1, z4, z4, z8, 0);
}

__device__ __forceinline__ unsigned lds_off_of(const void* p) {
    return (unsigned)(uintptr_t)p;   // low 32 bits of flat shared addr = LDS byte address
}

// ---------- problem constants ----------
#define GM    8192
#define GN    1024
#define GK    1024
#define SEQ   2048
#define HEADS 16
#define DK    64

// ---------- conversion kernels ----------
__global__ void conv_x_kernel(const float* __restrict__ in, bf16* __restrict__ out, int n) {
    for (int i = blockIdx.x * blockDim.x + threadIdx.x; i < n; i += gridDim.x * blockDim.x)
        out[i] = f2bf(in[i]);
}
__global__ void conv_wqkvT_kernel(const float* __restrict__ W, bf16* __restrict__ out) {
    const int total = GN * GK;
    for (int i = blockIdx.x * blockDim.x + threadIdx.x; i < total; i += gridDim.x * blockDim.x) {
        int n = i >> 10, d = i & 1023;
        out[i] = f2bf(W[(size_t)(n >> 6) * (1024 * 64) + (size_t)d * 64 + (n & 63)]);
    }
}
__global__ void conv_woT_kernel(const float* __restrict__ W, bf16* __restrict__ out) {
    const int total = GN * GK;
    for (int i = blockIdx.x * blockDim.x + threadIdx.x; i < total; i += gridDim.x * blockDim.x) {
        int n = i >> 10, m = i & 1023;
        out[i] = f2bf(W[(size_t)m * 1024 + n]);
    }
}

// ---------- NT GEMM with TDM double buffering ----------
// C[8192,1024] = A[8192,1024] * Bt[1024,1024]^T; block tile 128x128, k-step 64.
// STORE_MODE: 0 = f32 row-major, 1 = Q bf16 [b,h,s,dk] *0.125, 2 = K bf16, 3 = V bf16 [b,h,dk,s]
template <int STORE_MODE>
__global__ __launch_bounds__(256) void gemm_nt_kernel(const bf16* __restrict__ A,
                                                      const bf16* __restrict__ Bt,
                                                      void* __restrict__ Cout) {
    __shared__ __align__(16) bf16 lA[2][128 * 64];   // 2 x 16 KB
    __shared__ __align__(16) bf16 lB[2][128 * 64];   // 2 x 16 KB

    const int t = threadIdx.x, w = t >> 5, lane = t & 31;
    const int lg = lane >> 4, ln = lane & 15;
    const int wm = w & 1, wn = w >> 1;                 // 2 x 4 wave grid
    const int m0 = blockIdx.x * 128, n0 = blockIdx.y * 128;

    v8f acc[4][2];
#pragma unroll
    for (int i = 0; i < 4; ++i)
#pragma unroll
        for (int j = 0; j < 2; ++j) acc[i][j] = vzero8();

    const unsigned ldsA[2] = { lds_off_of(&lA[0][0]), lds_off_of(&lA[1][0]) };
    const unsigned ldsB[2] = { lds_off_of(&lB[0][0]), lds_off_of(&lB[1][0]) };

    constexpr int NIT = GK / 64;
    if (w == 0) {
        tdm_load_2d(ldsA[0], A + (size_t)m0 * GK, GK, GM, 64, 128, GK);
        tdm_load_2d(ldsB[0], Bt + (size_t)n0 * GK, GK, GN, 64, 128, GK);
    }
    for (int it = 0; it < NIT; ++it) {
        const int buf = it & 1;
        if (w == 0) __builtin_amdgcn_s_wait_tensorcnt((short)0);
        __syncthreads();                                   // publish buf, retire old reads
        if (w == 0 && it + 1 < NIT) {
            const int kk = (it + 1) * 64;
            tdm_load_2d(ldsA[buf ^ 1], A + (size_t)m0 * GK + kk, GK, GM, 64, 128, GK);
            tdm_load_2d(ldsB[buf ^ 1], Bt + (size_t)n0 * GK + kk, GK, GN, 64, 128, GK);
        }
        const bf16* As = &lA[buf][0];
        const bf16* Bs = &lB[buf][0];
#pragma unroll
        for (int kc = 0; kc < 2; ++kc) {
            V16 afr[4];
#pragma unroll
            for (int i = 0; i < 4; ++i) {
                const bf16* p = As + (wm * 64 + i * 16 + ln) * 64 + kc * 32;
                afr[i].h[0] = *(const v8bf*)(p + lg * 8);
                afr[i].h[1] = *(const v8bf*)(p + 16 + lg * 8);
            }
            V16 bfr[2];
#pragma unroll
            for (int j = 0; j < 2; ++j) {
                const bf16* p = Bs + (wn * 32 + j * 16 + ln) * 64 + kc * 32;
                bfr[j].h[0] = *(const v8bf*)(p + lg * 8);
                bfr[j].h[1] = *(const v8bf*)(p + 16 + lg * 8);
            }
#pragma unroll
            for (int i = 0; i < 4; ++i)
#pragma unroll
                for (int j = 0; j < 2; ++j)
                    acc[i][j] = wmma_bf16(afr[i].v, bfr[j].v, acc[i][j]);
        }
    }

#pragma unroll
    for (int i = 0; i < 4; ++i) {
#pragma unroll
        for (int j = 0; j < 2; ++j) {
#pragma unroll
            for (int r = 0; r < 8; ++r) {
                const int m = m0 + wm * 64 + i * 16 + lg * 8 + r;
                const int n = n0 + wn * 32 + j * 16 + ln;
                const float val = acc[i][j][r];
                if constexpr (STORE_MODE == 0) {
                    ((float*)Cout)[(size_t)m * GN + n] = val;
                } else {
                    const int b = m >> 11, s = m & 2047;
                    const int h = n >> 6,  d = n & 63;
                    if constexpr (STORE_MODE == 1) {
                        ((bf16*)Cout)[(((size_t)(b * HEADS + h) * SEQ + s) * DK) + d] =
                            f2bf(val * 0.125f);
                    } else if constexpr (STORE_MODE == 2) {
                        ((bf16*)Cout)[(((size_t)(b * HEADS + h) * SEQ + s) * DK) + d] = f2bf(val);
                    } else {
                        ((bf16*)Cout)[(((size_t)(b * HEADS + h) * DK + d) * SEQ) + s] = f2bf(val);
                    }
                }
            }
        }
    }
}

// ---------- flash attention: 64-key chunks, TDM-staged K/V, DPP softmax ----------
// grid (S/128, B*H); 8 waves; wave w owns 16 query rows; K/V chunks shared via LDS.
__global__ __launch_bounds__(256) void flash_attn_kernel(const bf16* __restrict__ Qb,
                                                         const bf16* __restrict__ Kb,
                                                         const bf16* __restrict__ Vt,
                                                         bf16* __restrict__ Ob) {
    __shared__ __align__(16) bf16 lK[2][64 * 64];    // 2 x 8 KB: rows = key, cols = dk
    __shared__ __align__(16) bf16 lV[2][64 * 64];    // 2 x 8 KB: rows = dk,  cols = key
    __shared__ __align__(16) bf16 lP[8][16 * 64];    // 16 KB: per-wave P staging

    const int t = threadIdx.x, w = t >> 5, lane = t & 31;
    const int lg = lane >> 4, ln = lane & 15;
    const int bh = blockIdx.y;
    const int q0 = blockIdx.x * 128 + w * 16;

    const bf16* Qh = Qb + (size_t)bh * SEQ * DK;
    const bf16* Kh = Kb + (size_t)bh * SEQ * DK;
    const bf16* Vh = Vt + (size_t)bh * DK * SEQ;

    V16 qa[2];
    {
        const bf16* qr = Qh + (size_t)(q0 + ln) * DK;
        qa[0].h[0] = *(const v8bf*)(qr + lg * 8);
        qa[0].h[1] = *(const v8bf*)(qr + 16 + lg * 8);
        qa[1].h[0] = *(const v8bf*)(qr + 32 + lg * 8);
        qa[1].h[1] = *(const v8bf*)(qr + 48 + lg * 8);
    }

    float mrun[8], lrun[8];
#pragma unroll
    for (int r = 0; r < 8; ++r) { mrun[r] = -3.0e38f; lrun[r] = 0.f; }
    v8f oacc[4];
#pragma unroll
    for (int j = 0; j < 4; ++j) oacc[j] = vzero8();

    const unsigned ldsK[2] = { lds_off_of(&lK[0][0]), lds_off_of(&lK[1][0]) };
    const unsigned ldsV[2] = { lds_off_of(&lV[0][0]), lds_off_of(&lV[1][0]) };

    constexpr int NIT = SEQ / 64;
    if (w == 0) {
        tdm_load_2d(ldsK[0], Kh, DK, SEQ, 64, 64, DK);            // K rows s0..s0+63
        tdm_load_2d(ldsV[0], Vh, SEQ, DK, 64, 64, SEQ);           // V cols s0..s0+63
    }
    for (int it = 0; it < NIT; ++it) {
        const int buf = it & 1;
        if (w == 0) __builtin_amdgcn_s_wait_tensorcnt((short)0);
        __syncthreads();
        if (w == 0 && it + 1 < NIT) {
            const int s1 = (it + 1) * 64;
            tdm_load_2d(ldsK[buf ^ 1], Kh + (size_t)s1 * DK, DK, SEQ, 64, 64, DK);
            tdm_load_2d(ldsV[buf ^ 1], Vh + s1, SEQ, DK, 64, 64, SEQ);
        }
        const bf16* Ks = &lK[buf][0];
        const bf16* Vs = &lV[buf][0];

        // ---- scores: 4 tiles of 16 keys, dk=64 as two 32-deep WMMA ----
        v8f sa[4];
#pragma unroll
        for (int jj = 0; jj < 4; ++jj) {
            const bf16* kr = Ks + (jj * 16 + ln) * 64;
            V16 k0, k1;
            k0.h[0] = *(const v8bf*)(kr + lg * 8);
            k0.h[1] = *(const v8bf*)(kr + 16 + lg * 8);
            k1.h[0] = *(const v8bf*)(kr + 32 + lg * 8);
            k1.h[1] = *(const v8bf*)(kr + 48 + lg * 8);
            v8f z = vzero8();
            z = wmma_bf16(qa[0].v, k0.v, z);
            z = wmma_bf16(qa[1].v, k1.v, z);
            sa[jj] = z;
        }

        // ---- online softmax over 64 keys (DPP reductions, no LDS) ----
#pragma unroll
        for (int r = 0; r < 8; ++r) {
            float v0 = sa[0][r], v1 = sa[1][r], v2 = sa[2][r], v3 = sa[3][r];
            float mx = fmaxf(fmaxf(v0, v1), fmaxf(v2, v3));
            mx = red_max16(mx);
            const float mnew  = fmaxf(mrun[r], mx);
            const float alpha = __expf(mrun[r] - mnew);
            const float p0 = __expf(v0 - mnew);
            const float p1 = __expf(v1 - mnew);
            const float p2 = __expf(v2 - mnew);
            const float p3 = __expf(v3 - mnew);
            float rs = (p0 + p1) + (p2 + p3);
            rs = red_sum16(rs);
            lrun[r] = lrun[r] * alpha + rs;
            mrun[r] = mnew;
#pragma unroll
            for (int j = 0; j < 4; ++j) oacc[j][r] *= alpha;
            const int m = lg * 8 + r;
            bf16* pb = &lP[w][m * 64 + ln];
            pb[0]  = f2bf(p0);
            pb[16] = f2bf(p1);
            pb[32] = f2bf(p2);
            pb[48] = f2bf(p3);
        }

        // ---- P (16x64) in A-operand layout from wave-private LDS ----
        V16 pa[2];
        {
            const bf16* pr = &lP[w][ln * 64];
            pa[0].h[0] = *(const v8bf*)(pr + lg * 8);
            pa[0].h[1] = *(const v8bf*)(pr + 16 + lg * 8);
            pa[1].h[0] = *(const v8bf*)(pr + 32 + lg * 8);
            pa[1].h[1] = *(const v8bf*)(pr + 48 + lg * 8);
        }
        // ---- O += P * V_chunk ----
#pragma unroll
        for (int j = 0; j < 4; ++j) {
            const bf16* vr = Vs + (j * 16 + ln) * 64;
            V16 b0, b1;
            b0.h[0] = *(const v8bf*)(vr + lg * 8);
            b0.h[1] = *(const v8bf*)(vr + 16 + lg * 8);
            b1.h[0] = *(const v8bf*)(vr + 32 + lg * 8);
            b1.h[1] = *(const v8bf*)(vr + 48 + lg * 8);
            oacc[j] = wmma_bf16(pa[0].v, b0.v, oacc[j]);
            oacc[j] = wmma_bf16(pa[1].v, b1.v, oacc[j]);
        }
    }

    // ---- finalize + store O as bf16 [b, s, h*64+dk] ----
    const int b = bh >> 4, h = bh & 15;
#pragma unroll
    for (int r = 0; r < 8; ++r) {
        const float inv = 1.0f / lrun[r];
        const int srow = q0 + lg * 8 + r;
        const size_t rowbase = ((size_t)(b * SEQ + srow)) * GN + h * DK;
#pragma unroll
        for (int j = 0; j < 4; ++j)
            Ob[rowbase + j * 16 + ln] = f2bf(oacc[j][r] * inv);
    }
}

// ---------- launcher ----------
extern "C" void kernel_launch(void* const* d_in, const int* in_sizes, int n_in,
                              void* d_out, int out_size, void* d_ws, size_t ws_size,
                              hipStream_t stream) {
    const float* x  = (const float*)d_in[0];
    const float* Wq = (const float*)d_in[1];
    const float* Wk = (const float*)d_in[2];
    const float* Wv = (const float*)d_in[3];
    const float* Wo = (const float*)d_in[4];

    char* ws = (char*)d_ws;
    const size_t MB = 1u << 20;
    bf16* xb  = (bf16*)(ws + 0 * MB);
    bf16* WqT = (bf16*)(ws + 16 * MB);
    bf16* WkT = (bf16*)(ws + 18 * MB);
    bf16* WvT = (bf16*)(ws + 20 * MB);
    bf16* WoT = (bf16*)(ws + 22 * MB);
    bf16* Qb  = (bf16*)(ws + 24 * MB);
    bf16* Kb  = (bf16*)(ws + 40 * MB);
    bf16* Vt  = (bf16*)(ws + 56 * MB);
    bf16* Ob  = (bf16*)(ws + 72 * MB);

    conv_x_kernel<<<4096, 256, 0, stream>>>(x, xb, GM * GK);
    conv_wqkvT_kernel<<<2048, 256, 0, stream>>>(Wq, WqT);
    conv_wqkvT_kernel<<<2048, 256, 0, stream>>>(Wk, WkT);
    conv_wqkvT_kernel<<<2048, 256, 0, stream>>>(Wv, WvT);
    conv_woT_kernel<<<2048, 256, 0, stream>>>(Wo, WoT);

    dim3 gg(GM / 128, GN / 128);
    gemm_nt_kernel<1><<<gg, 256, 0, stream>>>(xb, WqT, (void*)Qb);
    gemm_nt_kernel<2><<<gg, 256, 0, stream>>>(xb, WkT, (void*)Kb);
    gemm_nt_kernel<3><<<gg, 256, 0, stream>>>(xb, WvT, (void*)Vt);

    flash_attn_kernel<<<dim3(SEQ / 128, 4 * HEADS), 256, 0, stream>>>(Qb, Kb, Vt, Ob);

    gemm_nt_kernel<0><<<gg, 256, 0, stream>>>(Ob, WoT, d_out);
}
